// SparseSDFVQVAE_58987080843752
// MI455X (gfx1250) — matrix operand order repaired and checked by reading, not compiled, not run on hardware
//
#include <hip/hip_runtime.h>
#include <hip/hip_bf16.h>

// Problem constants (fixed by the reference).
constexpr int NN = 32768;   // rows of z
constexpr int KK = 8192;    // codebook size
constexpr int DD = 64;      // embedding dim
constexpr int NIT = KK / 64;         // K-chunks of 64 embeddings
constexpr int ES  = 80;              // eLds row stride in bf16 (conflict-free, 32B aligned)
constexpr int BUF_STRIDE = 10496;    // 64*80*2 (tile) + 256 (seed) bytes, 32B aligned
constexpr int ROWS_PER_BLOCK = 256;  // 8 waves x 32 rows (M=32 per wave: B-fragment reuse x2)

typedef __attribute__((ext_vector_type(16))) __bf16          v16bf;
typedef __attribute__((ext_vector_type(16))) unsigned short  v16us;
typedef __attribute__((ext_vector_type(8)))  float           v8f;

// float -> bf16 bits, round-to-nearest-even (bit trick; no __bf16 arithmetic needed)
__device__ __forceinline__ unsigned short f2bf(float f) {
    unsigned int u = __float_as_uint(f);
    unsigned int r = u + 0x7FFFu + ((u >> 16) & 1u);
    return (unsigned short)(r >> 16);
}

// ---------------------------------------------------------------------------
// Kernel 1: embeddings fp32 -> bf16 (row-major) + seed[k] = -0.5 * ||e_k||^2
// (seed pre-scaled so the main loop's WMMA C operand is a pure broadcast)
// ---------------------------------------------------------------------------
__global__ __launch_bounds__(256) void vq_prep(const float* __restrict__ emb,
                                               unsigned short* __restrict__ ebf,
                                               float* __restrict__ seedq) {
    const int tid = threadIdx.x;
    const int rl  = tid >> 6;
    const int d   = tid & 63;
    const int row = blockIdx.x * 4 + rl;
    const float v = emb[row * DD + d];
    ebf[row * DD + d] = f2bf(v);
    float s = v * v;
    #pragma unroll
    for (int m = 16; m >= 1; m >>= 1) s += __shfl_xor(s, m, 32);
    __shared__ float red[8];
    if ((tid & 31) == 0) red[tid >> 5] = s;
    __syncthreads();
    if (tid < 4) seedq[blockIdx.x * 4 + tid] = -0.5f * (red[2 * tid] + red[2 * tid + 1]);
}

// ---------------------------------------------------------------------------
// Kernel 2: main. grid = NN/256 blocks, 256 threads = 8 waves.
// Each wave owns 32 z rows (two 16x32 A-fragment sets) -> each B fragment
// loaded from LDS feeds two WMMAs. Double-buffered e-tiles, 1 barrier/chunk.
// Score = dot(z,e) - 0.5||e||^2 (seed as shared WMMA C operand); argmax of
// score == argmin of euclidean distance.
// ---------------------------------------------------------------------------
__global__ __launch_bounds__(256) void vq_main(const float* __restrict__ z,
                                               const float* __restrict__ emb,
                                               const unsigned short* __restrict__ ebf,
                                               const float* __restrict__ seedq,
                                               float* __restrict__ out,
                                               float* __restrict__ partial) {
    __shared__ alignas(32) char smemRaw[65536];
    float* zLds   = (float*)smemRaw;                 // phase 1: 256x64 f32 = 64KB
    int*   idxLds = (int*)(smemRaw + 21504);         // phase 3: 256 ints (disjoint from buffers)
    float* redLds = (float*)(smemRaw + 22528);

    const int tid  = threadIdx.x;
    const int wave = tid >> 5;
    const int lane = tid & 31;
    const int n    = lane & 15;     // column (B) / row (A) selector
    const int h    = lane >> 4;     // half-wave
    const int blockRow0 = blockIdx.x * ROWS_PER_BLOCK;
    const int rowc = tid >> 2, partc = tid & 3;      // cooperative-copy mapping

    // ---- stage z tile [256 x 64] fp32, coalesced float4 ----
    {
        const float4* zg = (const float4*)(z + (size_t)blockRow0 * DD);
        float4*       zl = (float4*)zLds;
        #pragma unroll
        for (int i = 0; i < 16; ++i) zl[tid + 256 * i] = zg[tid + 256 * i];
    }

    // ---- preload K-chunk 0 into registers (overlaps with z staging) ----
    uint4 p0, p1; float pe;
    {
        const uint4* src = (const uint4*)ebf;
        p0 = src[rowc * 8 + partc * 2];
        p1 = src[rowc * 8 + partc * 2 + 1];
        pe = (tid < 64) ? seedq[tid] : 0.0f;
    }
    __syncthreads();

    // ---- build A fragments: two row sets (m = n and m = n+16 within the wave's 32) ----
    // elem j -> k = 16h+j  (+32 for second K-chunk)
    v16bf a[2][2];   // [rowset][kchunk]
    #pragma unroll
    for (int s = 0; s < 2; ++s) {
        const float* zr = zLds + (wave * 32 + s * 16 + n) * DD;
        v16us u0, u1;
        #pragma unroll
        for (int j = 0; j < 8; ++j) {
            u0[j]     = f2bf(zr[      16 * h + j]);
            u0[j + 8] = f2bf(zr[8   + 16 * h + j]);
            u1[j]     = f2bf(zr[32  + 16 * h + j]);
            u1[j + 8] = f2bf(zr[40  + 16 * h + j]);
        }
        a[s][0] = __builtin_bit_cast(v16bf, u0);
        a[s][1] = __builtin_bit_cast(v16bf, u1);
    }
    __syncthreads();   // zLds reads done in ALL waves before buffer stores alias it

    float bestVal[2][8], bestIdx[2][8];
    #pragma unroll
    for (int s = 0; s < 2; ++s)
        #pragma unroll
        for (int r = 0; r < 8; ++r) { bestVal[s][r] = -3.402823e38f; bestIdx[s][r] = 0.0f; }

    // ---- K loop: 64 embeddings per chunk, LDS double-buffered ----
    for (int it = 0; it < NIT; ++it) {
        unsigned short* eL  = (unsigned short*)(smemRaw + (it & 1) * BUF_STRIDE);
        float*          eqL = (float*)(smemRaw + (it & 1) * BUF_STRIDE + 64 * ES * 2);

        {   // store preloaded chunk (stride-80 rows), 32B per thread
            uint4* dst = (uint4*)eL;
            dst[rowc * 10 + partc * 2]     = p0;
            dst[rowc * 10 + partc * 2 + 1] = p1;
            if (tid < 64) eqL[tid] = pe;
        }
        if (it + 1 < NIT) {   // next chunk's global loads hidden under compute
            const uint4* src = (const uint4*)(ebf + (size_t)(it + 1) * 64 * DD);
            p0 = src[rowc * 8 + partc * 2];
            p1 = src[rowc * 8 + partc * 2 + 1];
            if (tid < 64) pe = seedq[(it + 1) * 64 + tid];
        }
        __syncthreads();

        // one broadcast seed per column tile, shared by both row sets as WMMA C
        v8f acc[2][4];
        #pragma unroll
        for (int t = 0; t < 4; ++t) {
            const float eq = eqL[t * 16 + n];        // already -0.5*||e||^2
            v8f seed;
            #pragma unroll
            for (int r = 0; r < 8; ++r) seed[r] = eq;
            v16bf b0 = *(const v16bf*)(const void*)(eL + (t * 16 + n) * ES + 16 * h);
            acc[0][t] = __builtin_amdgcn_wmma_f32_16x16x32_bf16(false, a[0][0], false, b0,
                                                                (short)0, seed, false, false);
            acc[1][t] = __builtin_amdgcn_wmma_f32_16x16x32_bf16(false, a[1][0], false, b0,
                                                                (short)0, seed, false, false);
        }
        #pragma unroll
        for (int t = 0; t < 4; ++t) {
            v16bf b1 = *(const v16bf*)(const void*)(eL + (t * 16 + n) * ES + 32 + 16 * h);
            acc[0][t] = __builtin_amdgcn_wmma_f32_16x16x32_bf16(false, a[0][1], false, b1,
                                                                (short)0, acc[0][t], false, false);
            acc[1][t] = __builtin_amdgcn_wmma_f32_16x16x32_bf16(false, a[1][1], false, b1,
                                                                (short)0, acc[1][t], false, false);
        }
        // batched running argmax (first-wins on ties; per-lane columns ascend with it)
        const int kBase = it * 64;
        #pragma unroll
        for (int t = 0; t < 4; ++t) {
            const float colF = (float)(kBase + t * 16 + n);
            #pragma unroll
            for (int s = 0; s < 2; ++s)
                #pragma unroll
                for (int r = 0; r < 8; ++r) {
                    const bool better = acc[s][t][r] > bestVal[s][r];
                    bestIdx[s][r] = better ? colF         : bestIdx[s][r];
                    bestVal[s][r] = better ? acc[s][t][r] : bestVal[s][r];
                }
        }
    }

    // ---- reduce argmax across the 16-lane column group (xor < 16 stays in-group) ----
    #pragma unroll
    for (int mask = 1; mask <= 8; mask <<= 1) {
        #pragma unroll
        for (int s = 0; s < 2; ++s)
            #pragma unroll
            for (int r = 0; r < 8; ++r) {
                const float ov = __shfl_xor(bestVal[s][r], mask, 32);
                const float oi = __shfl_xor(bestIdx[s][r], mask, 32);
                if (ov > bestVal[s][r] || (ov == bestVal[s][r] && oi < bestIdx[s][r])) {
                    bestVal[s][r] = ov; bestIdx[s][r] = oi;
                }
            }
    }
    if (n == 0) {   // lanes 0 and 16: rows m = s*16 + 8h + r
        #pragma unroll
        for (int s = 0; s < 2; ++s)
            #pragma unroll
            for (int r = 0; r < 8; ++r)
                idxLds[wave * 32 + s * 16 + 8 * h + r] = (int)bestIdx[s][r];
    }
    __syncthreads();

    // ---- indices output (float), offset N*D + 2 ----
    out[(size_t)NN * DD + 2 + blockRow0 + tid] = (float)idxLds[tid];

    // ---- gather winning embeddings, write quantized, accumulate loss ----
    float sum = 0.0f;
    #pragma unroll
    for (int i = 0; i < 64; ++i) {
        const int idx = tid + 256 * i;          // 0..16383 over the 256x64 tile
        const int row = idx >> 6, d = idx & 63;
        const int kk  = idxLds[row];
        const float q  = emb[(size_t)kk * DD + d];
        const float zv = z[(size_t)(blockRow0 + row) * DD + d];
        out[(size_t)(blockRow0 + row) * DD + d] = q;
        const float diff = q - zv;
        sum += diff * diff;
    }
    #pragma unroll
    for (int m = 16; m >= 1; m >>= 1) sum += __shfl_xor(sum, m, 32);
    if (lane == 0) redLds[wave] = sum;
    __syncthreads();
    if (tid == 0) {
        float s = 0.0f;
        #pragma unroll
        for (int w = 0; w < 8; ++w) s += redLds[w];
        partial[blockIdx.x] = s;                 // deterministic (no atomics)
    }
}

// ---------------------------------------------------------------------------
// Kernel 3: deterministic reduction of 128 block partials -> both losses
// ---------------------------------------------------------------------------
__global__ __launch_bounds__(256) void vq_finalize(const float* __restrict__ partial,
                                                   float* __restrict__ out) {
    __shared__ float s[256];
    const int tid = threadIdx.x;
    s[tid] = (tid < NN / ROWS_PER_BLOCK) ? partial[tid] : 0.0f;
    __syncthreads();
    for (int st = 128; st > 0; st >>= 1) {
        if (tid < st) s[tid] += s[tid + st];
        __syncthreads();
    }
    if (tid == 0) {
        const float mean = s[0] / (float)((size_t)NN * DD);
        out[(size_t)NN * DD]     = mean;   // vq_loss
        out[(size_t)NN * DD + 1] = mean;   // commitment_loss (numerically identical)
    }
}

// ---------------------------------------------------------------------------
extern "C" void kernel_launch(void* const* d_in, const int* in_sizes, int n_in,
                              void* d_out, int out_size, void* d_ws, size_t ws_size,
                              hipStream_t stream) {
    (void)in_sizes; (void)n_in; (void)out_size; (void)ws_size;
    const float* z   = (const float*)d_in[0];   // [N, D]
    const float* emb = (const float*)d_in[1];   // [K, D]
    float* out = (float*)d_out;

    char* ws = (char*)d_ws;                     // workspace layout (~1.1 MB):
    unsigned short* ebf = (unsigned short*)ws;                              // K*D bf16
    float* seedq   = (float*)(ws + (size_t)KK * DD * 2);                    // K f32: -0.5*||e||^2
    float* partial = (float*)(ws + (size_t)KK * DD * 2 + (size_t)KK * 4);   // 128 f32

    vq_prep    <<<KK / 4,              256, 0, stream>>>(emb, ebf, seedq);
    vq_main    <<<NN / ROWS_PER_BLOCK, 256, 0, stream>>>(z, emb, ebf, seedq, out, partial);
    vq_finalize<<<1,                   256, 0, stream>>>(partial, out);
}
